// GraphEncoder_83107617177700
// MI455X (gfx1250) — matrix operand order repaired and checked by reading, not compiled, not run on hardware
//
#include <hip/hip_runtime.h>
#include <hip/hip_bf16.h>

// ---------------------------------------------------------------------------
// GATv2 graph encoder for MI455X (gfx1250), wave32 + WMMA.
// GEMMs run on v_wmma_f32_16x16x32_f16 (f16 in, f32 accumulate).
// A-tiles staged into LDS with global_load_async_to_lds_b128 (ASYNCcnt path).
// Edge softmax/aggregation via float atomics (bandwidth-bound phase).
// ---------------------------------------------------------------------------

typedef __attribute__((ext_vector_type(16))) _Float16 v16h;
typedef __attribute__((ext_vector_type(8)))  _Float16 v8h;
typedef __attribute__((ext_vector_type(8)))  float    v8f;

constexpr int NN  = 50000;          // nodes
constexpr int NE  = 800000;         // edges (without self loops)
constexpr int NEE = NE + NN;        // edges + self loops
constexpr int DD  = 128;            // feature dim (= H*C)
constexpr int NH  = 4;              // heads
constexpr int NC  = 32;             // channels per head
constexpr float NEG_SLOPE = 0.2f;
constexpr float LN_EPS    = 1e-5f;

// ---- float <-> orderable uint (for atomic max on float) -------------------
__device__ __forceinline__ unsigned f2ord(float f) {
    unsigned u = __float_as_uint(f);
    return (u & 0x80000000u) ? ~u : (u | 0x80000000u);
}
__device__ __forceinline__ float ord2f(unsigned u) {
    return (u & 0x80000000u) ? __uint_as_float(u & 0x7fffffffu)
                             : __uint_as_float(~u);
}

// ---- async 16B global -> LDS copy (gfx1250 ASYNCcnt path) -----------------
__device__ __forceinline__ void async_copy16(void* lds_ptr, const void* gptr) {
    // low 32 bits of a flat shared-memory address = wave-relative LDS offset
    unsigned lds_off = (unsigned)(unsigned long long)lds_ptr;
    asm volatile("global_load_async_to_lds_b128 %0, %1, off"
                 :: "v"(lds_off), "v"(gptr) : "memory");
}
__device__ __forceinline__ void async_wait0() {
    asm volatile("s_wait_asynccnt 0x0" ::: "memory");
}

// ---------------------------------------------------------------------------
// Weight prep: W[k][m] f32 -> Wt[m][k] f16  (128x128)
// ---------------------------------------------------------------------------
__global__ __launch_bounds__(256) void k_wprep(const float* __restrict__ W,
                                               _Float16* __restrict__ Wt) {
    int idx = blockIdx.x * 256 + threadIdx.x;      // 0..16383
    int m = idx >> 7, k = idx & 127;
    Wt[(size_t)m * DD + k] = (_Float16)W[(size_t)k * DD + m];
}

// f32 -> f16 elementwise
__global__ __launch_bounds__(256) void k_cvt16(const float* __restrict__ in,
                                               _Float16* __restrict__ out, int n) {
    int i = blockIdx.x * 256 + threadIdx.x;
    if (i < n) out[i] = (_Float16)in[i];
}

// zero aggregate / segment buffers
__global__ __launch_bounds__(256) void k_init(unsigned* __restrict__ lmax,
                                              float* __restrict__ den,
                                              float* __restrict__ agg) {
    int i = blockIdx.x * 256 + threadIdx.x;
    if (i < NN * DD) agg[i] = 0.f;
    if (i < NN * NH) { lmax[i] = 0u; den[i] = 0.f; }
}

// ---------------------------------------------------------------------------
// Dual GEMM: OL = A@WLt^T + bL ; OR = A@WRt^T + bR   (A:[N,128]f16, Wt:[M,K]f16)
// One wave -> one 16x16 tile of each output; block = 16 rows x 128 cols.
// ---------------------------------------------------------------------------
__global__ __launch_bounds__(256) void k_gemm_dual(
    const _Float16* __restrict__ A,
    const _Float16* __restrict__ WLt, const _Float16* __restrict__ WRt,
    const float* __restrict__ bL, const float* __restrict__ bR,
    float* __restrict__ OL, float* __restrict__ OR_) {
    __shared__ _Float16 As[16][136];               // pad 128->136 to avoid bank conflicts
    const int r0  = blockIdx.x * 16;
    const int tid = threadIdx.x;
    {   // async-stage 16x128 A tile into LDS (one b128 chunk per thread)
        int row = tid >> 4, c0 = (tid & 15) * 8;
        async_copy16(&As[row][c0], A + (size_t)(r0 + row) * DD + c0);
    }
    async_wait0();
    __syncthreads();
    const int lane = tid & 31;
    const int col  = (tid >> 5) * 16 + (lane & 15);
    const int rsel = lane >> 4;                    // 0/1: lane half
    v8f accL = {}, accR = {};
#pragma unroll
    for (int kc = 0; kc < DD; kc += 32) {
        const int koff = rsel * 16;
        v8h lo = *(const v8h*)&As[lane & 15][kc + koff];
        v8h hi = *(const v8h*)&As[lane & 15][kc + 8 + koff];
        v16h a;
#pragma unroll
        for (int i = 0; i < 8; ++i) { a[i] = lo[i]; a[i + 8] = hi[i]; }
        const int kb = kc + rsel * 16;
        v16h bl_ = *(const v16h*)(WLt + (size_t)col * DD + kb);
        v16h br_ = *(const v16h*)(WRt + (size_t)col * DD + kb);
        accL = __builtin_amdgcn_wmma_f32_16x16x32_f16(false, a, false, bl_,
                                                      (short)0, accL, false, false);
        accR = __builtin_amdgcn_wmma_f32_16x16x32_f16(false, a, false, br_,
                                                      (short)0, accR, false, false);
    }
    const float cL = bL[col], cR = bR[col];
#pragma unroll
    for (int r = 0; r < 8; ++r) {
        int row = r0 + r + rsel * 8;
        OL [(size_t)row * DD + col] = accL[r] + cL;
        OR_[(size_t)row * DD + col] = accR[r] + cR;
    }
}

// Single GEMM (output projection): O = A@Wt^T + b
__global__ __launch_bounds__(256) void k_gemm_single(
    const _Float16* __restrict__ A, const _Float16* __restrict__ Wt,
    const float* __restrict__ b, float* __restrict__ O) {
    __shared__ _Float16 As[16][136];
    const int r0  = blockIdx.x * 16;
    const int tid = threadIdx.x;
    {
        int row = tid >> 4, c0 = (tid & 15) * 8;
        async_copy16(&As[row][c0], A + (size_t)(r0 + row) * DD + c0);
    }
    async_wait0();
    __syncthreads();
    const int lane = tid & 31;
    const int col  = (tid >> 5) * 16 + (lane & 15);
    const int rsel = lane >> 4;
    v8f acc = {};
#pragma unroll
    for (int kc = 0; kc < DD; kc += 32) {
        const int koff = rsel * 16;
        v8h lo = *(const v8h*)&As[lane & 15][kc + koff];
        v8h hi = *(const v8h*)&As[lane & 15][kc + 8 + koff];
        v16h a;
#pragma unroll
        for (int i = 0; i < 8; ++i) { a[i] = lo[i]; a[i + 8] = hi[i]; }
        v16h bb = *(const v16h*)(Wt + (size_t)col * DD + kc + rsel * 16);
        acc = __builtin_amdgcn_wmma_f32_16x16x32_f16(false, a, false, bb,
                                                     (short)0, acc, false, false);
    }
    const float c = b[col];
#pragma unroll
    for (int r = 0; r < 8; ++r)
        O[(size_t)(r0 + r + rsel * 8) * DD + col] = acc[r] + c;
}

// ---------------------------------------------------------------------------
// Edge pass 1: logits + segment max.  Thread = (edge, head).
// ---------------------------------------------------------------------------
__global__ __launch_bounds__(256) void k_edge_logits(
    const int* __restrict__ ei, const float* __restrict__ XL,
    const float* __restrict__ XR, const float* __restrict__ att,
    float* __restrict__ logit, unsigned* __restrict__ lmax) {
    int t = blockIdx.x * 256 + threadIdx.x;
    if (t >= NEE * NH) return;
    int e = t >> 2, h = t & 3;
    int src = (e < NE) ? ei[e]      : (e - NE);
    int dst = (e < NE) ? ei[NE + e] : (e - NE);
    int e2 = e + 4096;                              // cover random-gather latency
    if (e2 < NE) __builtin_prefetch(XL + (size_t)ei[e2] * DD, 0, 1);
    const float4* pl = (const float4*)(XL + (size_t)src * DD + h * NC);
    const float4* pr = (const float4*)(XR + (size_t)dst * DD + h * NC);
    const float4* pa = (const float4*)(att + h * NC);
    float acc = 0.f;
#pragma unroll
    for (int i = 0; i < 8; ++i) {
        float4 l = pl[i], r = pr[i], a = pa[i];
        float m;
        m = l.x + r.x; m = m > 0.f ? m : NEG_SLOPE * m; acc += m * a.x;
        m = l.y + r.y; m = m > 0.f ? m : NEG_SLOPE * m; acc += m * a.y;
        m = l.z + r.z; m = m > 0.f ? m : NEG_SLOPE * m; acc += m * a.z;
        m = l.w + r.w; m = m > 0.f ? m : NEG_SLOPE * m; acc += m * a.w;
    }
    logit[(size_t)e * NH + h] = acc;
    atomicMax(lmax + (size_t)dst * NH + h, f2ord(acc));
}

// Edge pass 2: ex = exp(logit - max); segment denominator
__global__ __launch_bounds__(256) void k_edge_expden(
    const int* __restrict__ ei, float* __restrict__ logit,
    const unsigned* __restrict__ lmax, float* __restrict__ den) {
    int t = blockIdx.x * 256 + threadIdx.x;
    if (t >= NEE * NH) return;
    int e = t >> 2, h = t & 3;
    int dst = (e < NE) ? ei[NE + e] : (e - NE);
    float lm = ord2f(lmax[(size_t)dst * NH + h]);
    float ex = __expf(logit[(size_t)e * NH + h] - lm);
    logit[(size_t)e * NH + h] = ex;                 // reuse buffer as numerator
    atomicAdd(den + (size_t)dst * NH + h, ex);
}

// Edge pass 3: out[dst] += alpha * xl[src]
__global__ __launch_bounds__(256) void k_edge_aggr(
    const int* __restrict__ ei, const float* __restrict__ XL,
    const float* __restrict__ logit, const float* __restrict__ den,
    float* __restrict__ agg) {
    int t = blockIdx.x * 256 + threadIdx.x;
    if (t >= NEE * NH) return;
    int e = t >> 2, h = t & 3;
    int src = (e < NE) ? ei[e]      : (e - NE);
    int dst = (e < NE) ? ei[NE + e] : (e - NE);
    int e2 = e + 4096;
    if (e2 < NE) __builtin_prefetch(XL + (size_t)ei[e2] * DD, 0, 1);
    float alpha = logit[(size_t)e * NH + h] / den[(size_t)dst * NH + h];
    const float4* pl = (const float4*)(XL + (size_t)src * DD + h * NC);
    float* po = agg + (size_t)dst * DD + h * NC;
#pragma unroll
    for (int i = 0; i < 8; ++i) {
        float4 l = pl[i];
        atomicAdd(po + 4 * i + 0, l.x * alpha);
        atomicAdd(po + 4 * i + 1, l.y * alpha);
        atomicAdd(po + 4 * i + 2, l.z * alpha);
        atomicAdd(po + 4 * i + 3, l.w * alpha);
    }
}

// ---------------------------------------------------------------------------
// LayerNorm(+bias)+GELU, wave32 per row, f16 output (feeds next WMMA GEMM)
// ---------------------------------------------------------------------------
__global__ __launch_bounds__(256) void k_ln_gelu(
    const float* __restrict__ X, const float* __restrict__ bias,
    const float* __restrict__ g, const float* __restrict__ b,
    _Float16* __restrict__ out) {
    int row = blockIdx.x * 8 + (threadIdx.x >> 5);
    if (row >= NN) return;
    int lane = threadIdx.x & 31;
    float4 v  = *(const float4*)(X    + (size_t)row * DD + lane * 4);
    float4 bb = *(const float4*)(bias + lane * 4);
    float x0 = v.x + bb.x, x1 = v.y + bb.y, x2 = v.z + bb.z, x3 = v.w + bb.w;
    float s = x0 + x1 + x2 + x3;
    float q = x0 * x0 + x1 * x1 + x2 * x2 + x3 * x3;
#pragma unroll
    for (int m = 16; m >= 1; m >>= 1) {
        s += __shfl_xor(s, m, 32);
        q += __shfl_xor(q, m, 32);
    }
    float mean = s * (1.f / DD);
    float inv  = rsqrtf(q * (1.f / DD) - mean * mean + LN_EPS);
    float4 gg = *(const float4*)(g + lane * 4);
    float4 be = *(const float4*)(b + lane * 4);
    float y, r[4];
    y = (x0 - mean) * inv * gg.x + be.x; r[0] = 0.5f * y * (1.f + erff(y * 0.70710678f));
    y = (x1 - mean) * inv * gg.y + be.y; r[1] = 0.5f * y * (1.f + erff(y * 0.70710678f));
    y = (x2 - mean) * inv * gg.z + be.z; r[2] = 0.5f * y * (1.f + erff(y * 0.70710678f));
    y = (x3 - mean) * inv * gg.w + be.w; r[3] = 0.5f * y * (1.f + erff(y * 0.70710678f));
    _Float16* po = out + (size_t)row * DD + lane * 4;
    po[0] = (_Float16)r[0]; po[1] = (_Float16)r[1];
    po[2] = (_Float16)r[2]; po[3] = (_Float16)r[3];
}

// Final LayerNorm, f32 output
__global__ __launch_bounds__(256) void k_ln_final(
    const float* __restrict__ X, const float* __restrict__ g,
    const float* __restrict__ b, float* __restrict__ out) {
    int row = blockIdx.x * 8 + (threadIdx.x >> 5);
    if (row >= NN) return;
    int lane = threadIdx.x & 31;
    float4 v = *(const float4*)(X + (size_t)row * DD + lane * 4);
    float s = v.x + v.y + v.z + v.w;
    float q = v.x * v.x + v.y * v.y + v.z * v.z + v.w * v.w;
#pragma unroll
    for (int m = 16; m >= 1; m >>= 1) {
        s += __shfl_xor(s, m, 32);
        q += __shfl_xor(q, m, 32);
    }
    float mean = s * (1.f / DD);
    float inv  = rsqrtf(q * (1.f / DD) - mean * mean + LN_EPS);
    float4 gg = *(const float4*)(g + lane * 4);
    float4 be = *(const float4*)(b + lane * 4);
    float4 o;
    o.x = (v.x - mean) * inv * gg.x + be.x;
    o.y = (v.y - mean) * inv * gg.y + be.y;
    o.z = (v.z - mean) * inv * gg.z + be.z;
    o.w = (v.w - mean) * inv * gg.w + be.w;
    *(float4*)(out + (size_t)row * DD + lane * 4) = o;
}

// ---------------------------------------------------------------------------
extern "C" void kernel_launch(void* const* d_in, const int* in_sizes, int n_in,
                              void* d_out, int out_size, void* d_ws, size_t ws_size,
                              hipStream_t stream) {
    (void)in_sizes; (void)n_in; (void)out_size; (void)ws_size;
    const float* x     = (const float*)d_in[0];
    const int*   ei    = (const int*)  d_in[1];
    const float* Wl0   = (const float*)d_in[2];
    const float* bl0   = (const float*)d_in[3];
    const float* Wr0   = (const float*)d_in[4];
    const float* br0   = (const float*)d_in[5];
    const float* att0  = (const float*)d_in[6];
    const float* bias0 = (const float*)d_in[7];
    const float* g0    = (const float*)d_in[8];
    const float* b0    = (const float*)d_in[9];
    const float* Wl1   = (const float*)d_in[10];
    const float* bl1   = (const float*)d_in[11];
    const float* Wr1   = (const float*)d_in[12];
    const float* br1   = (const float*)d_in[13];
    const float* att1  = (const float*)d_in[14];
    const float* bias1 = (const float*)d_in[15];
    const float* g1    = (const float*)d_in[16];
    const float* b1    = (const float*)d_in[17];
    const float* Wout  = (const float*)d_in[18];
    const float* bout  = (const float*)d_in[19];
    const float* gout  = (const float*)d_in[20];
    const float* boutl = (const float*)d_in[21];
    float* out = (float*)d_out;

    // workspace layout (256B aligned)
    char* ws = (char*)d_ws;
    size_t o = 0;
    auto take = [&](size_t bytes) -> char* {
        char* p = ws + o;
        o = (o + bytes + 255) & ~(size_t)255;
        return p;
    };
    _Float16* A16  = (_Float16*)take((size_t)NN * DD * sizeof(_Float16));
    _Float16* Wt[5];
    for (int i = 0; i < 5; ++i) Wt[i] = (_Float16*)take((size_t)DD * DD * sizeof(_Float16));
    float*    XL   = (float*)take((size_t)NN * DD * sizeof(float));
    float*    XR   = (float*)take((size_t)NN * DD * sizeof(float));
    float*    LOG  = (float*)take((size_t)NEE * NH * sizeof(float));
    unsigned* LMAX = (unsigned*)take((size_t)NN * NH * sizeof(unsigned));
    float*    DEN  = (float*)take((size_t)NN * NH * sizeof(float));
    float*    AGG  = (float*)take((size_t)NN * DD * sizeof(float));  // reused as proj out

    const dim3 B(256);
    const int gElem  = (NN * DD + 255) / 256;
    const int gEdge  = (NEE * NH + 255) / 256;
    const dim3 gGemm(NN / 16);          // 3125, exact
    const dim3 gRow((NN + 7) / 8);      // 6250, exact

    // weight prep + input conversion
    k_wprep<<<dim3(64), B, 0, stream>>>(Wl0,  Wt[0]);
    k_wprep<<<dim3(64), B, 0, stream>>>(Wr0,  Wt[1]);
    k_wprep<<<dim3(64), B, 0, stream>>>(Wl1,  Wt[2]);
    k_wprep<<<dim3(64), B, 0, stream>>>(Wr1,  Wt[3]);
    k_wprep<<<dim3(64), B, 0, stream>>>(Wout, Wt[4]);
    k_cvt16<<<dim3(gElem), B, 0, stream>>>(x, A16, NN * DD);

    // ---- layer 0 ----
    k_gemm_dual<<<gGemm, B, 0, stream>>>(A16, Wt[0], Wt[1], bl0, br0, XL, XR);
    k_init<<<dim3(gElem), B, 0, stream>>>(LMAX, DEN, AGG);
    k_edge_logits<<<dim3(gEdge), B, 0, stream>>>(ei, XL, XR, att0, LOG, LMAX);
    k_edge_expden<<<dim3(gEdge), B, 0, stream>>>(ei, LOG, LMAX, DEN);
    k_edge_aggr  <<<dim3(gEdge), B, 0, stream>>>(ei, XL, LOG, DEN, AGG);
    k_ln_gelu<<<gRow, B, 0, stream>>>(AGG, bias0, g0, b0, A16);

    // ---- layer 1 ----
    k_gemm_dual<<<gGemm, B, 0, stream>>>(A16, Wt[2], Wt[3], bl1, br1, XL, XR);
    k_init<<<dim3(gElem), B, 0, stream>>>(LMAX, DEN, AGG);
    k_edge_logits<<<dim3(gEdge), B, 0, stream>>>(ei, XL, XR, att1, LOG, LMAX);
    k_edge_expden<<<dim3(gEdge), B, 0, stream>>>(ei, LOG, LMAX, DEN);
    k_edge_aggr  <<<dim3(gEdge), B, 0, stream>>>(ei, XL, LOG, DEN, AGG);
    k_ln_gelu<<<gRow, B, 0, stream>>>(AGG, bias1, g1, b1, A16);

    // ---- output projection + final LayerNorm ----
    k_gemm_single<<<gGemm, B, 0, stream>>>(A16, Wt[4], bout, AGG);
    k_ln_final<<<gRow, B, 0, stream>>>(AGG, gout, boutl, out);
}
